// Gemma4TextAttention_60979945669120
// MI455X (gfx1250) — compile-verified, matrix operand order checked by reading, not run on hardware
//
#include <hip/hip_runtime.h>
#include <hip/hip_bf16.h>
#include <math.h>

// ---------------------------------------------------------------------------
// Gemma-style attention for MI455X (gfx1250), wave32 + WMMA bf16 16x16x32.
// Pipeline: convert/pack(T) -> QKV GEMM -> norm+rope -> flash attn -> O GEMM
// CDNA5 paths: v_wmma_f32_16x16x32_bf16, global_prefetch_b8,
//              global_load_async_to_lds_b128 + s_wait_asynccnt.
// ---------------------------------------------------------------------------

typedef __attribute__((ext_vector_type(16))) __bf16 v16bf;
typedef __attribute__((ext_vector_type(8)))  __bf16 v8bf;
typedef __attribute__((ext_vector_type(8)))  float  v8f;

#define B_SZ   2
#define SEQ    2048
#define DMODEL 2048
#define HD     256
#define NQH    8
#define NKVH   4
#define MROWS  (B_SZ * SEQ)            // 4096
#define NCOLS  ((NQH + 2 * NKVH) * HD) // 4096
#define NEG_BIG (-1.0e9f)

static __device__ inline v16bf make_frag(const __bf16* p0, const __bf16* p1) {
    v8bf a = *(const v8bf*)p0;
    v8bf b = *(const v8bf*)p1;
    v16bf r;
#pragma unroll
    for (int i = 0; i < 8; ++i) { r[i] = a[i]; r[i + 8] = b[i]; }
    return r;
}

// async global->LDS 16B copy (CDNA5, tracked by ASYNCcnt)
static __device__ inline void async_copy_b128(const __bf16* g, __bf16* lds) {
    unsigned l = (unsigned)(size_t)lds;          // low 32 bits = LDS offset
    unsigned long long ga = (unsigned long long)(size_t)g;
    asm volatile("global_load_async_to_lds_b128 %0, %1, off"
                 :: "v"(l), "v"(ga) : "memory");
}

static __device__ inline void wait_asynccnt0() {
#if defined(__has_builtin)
#if __has_builtin(__builtin_amdgcn_s_wait_asynccnt)
    __builtin_amdgcn_s_wait_asynccnt(0);
    return;
#else
    asm volatile("s_wait_asynccnt 0" ::: "memory");
    return;
#endif
#else
    asm volatile("s_wait_asynccnt 0" ::: "memory");
#endif
}

// ---------------------------------------------------------------------------
// Kernel 1: fp32 -> bf16 elementwise convert
// ---------------------------------------------------------------------------
__global__ __launch_bounds__(256) void f32_to_bf16_kernel(const float* __restrict__ in,
                                                          __bf16* __restrict__ out, int n) {
    int i = blockIdx.x * 256 + threadIdx.x;
    if (i < n) out[i] = (__bf16)in[i];
}

// ---------------------------------------------------------------------------
// Kernel 2a: pack Wq/Wk/Wv [head][D][HD] fp32 -> bf16 TRANSPOSED [N=4096][K=D]
// column layout along N: [ q0..q7 | k0..k3 | v0..v3 ] * 256
// ---------------------------------------------------------------------------
__global__ __launch_bounds__(256) void pack_qkv_wT_kernel(const float* __restrict__ Wq,
                                                          const float* __restrict__ Wk,
                                                          const float* __restrict__ Wv,
                                                          __bf16* __restrict__ out) {
    int i = blockIdx.x * 256 + threadIdx.x;   // i over NCOLS * DMODEL, k fastest
    int k   = i & (DMODEL - 1);
    int col = i >> 11;
    float v;
    if (col < NQH * HD) {
        int head = col >> 8, h = col & 255;
        v = Wq[(size_t)head * DMODEL * HD + (size_t)k * HD + h];
    } else if (col < (NQH + NKVH) * HD) {
        int c = col - NQH * HD; int head = c >> 8, h = c & 255;
        v = Wk[(size_t)head * DMODEL * HD + (size_t)k * HD + h];
    } else {
        int c = col - (NQH + NKVH) * HD; int head = c >> 8, h = c & 255;
        v = Wv[(size_t)head * DMODEL * HD + (size_t)k * HD + h];
    }
    out[(size_t)col * DMODEL + k] = (__bf16)v;
}

// ---------------------------------------------------------------------------
// Kernel 2b: Wo flat [K=2048][D] fp32 -> bf16 TRANSPOSED [N=D][K=2048]
// ---------------------------------------------------------------------------
__global__ __launch_bounds__(256) void pack_woT_kernel(const float* __restrict__ Wo,
                                                       __bf16* __restrict__ out) {
    int i = blockIdx.x * 256 + threadIdx.x;   // i over D*2048, k fastest
    int k = i & 2047;
    int d = i >> 11;
    out[(size_t)d * 2048 + k] = (__bf16)Wo[(size_t)k * DMODEL + d];
}

// ---------------------------------------------------------------------------
// Kernel 3: bf16 GEMM, C(f32)[M][N] = A[M][K] * Bt[N][K]^T
// block tile 128x128, K-step 64, 8 waves (2x4), each wave 64x32 (4x2 wmma)
// ---------------------------------------------------------------------------
#define GT_BM 128
#define GT_BN 128
#define GT_BK 64

__global__ __launch_bounds__(256) void gemm_bf16_kernel(const __bf16* __restrict__ A,
                                                        const __bf16* __restrict__ Bt,
                                                        float* __restrict__ C,
                                                        int M, int N, int K) {
    __shared__ __bf16 sA[GT_BM][GT_BK];   // row-major [m][k]   (16 KB)
    __shared__ __bf16 sB[GT_BN][GT_BK];   // row-major [n][k]   (16 KB)

    const int bm = blockIdx.y * GT_BM;
    const int bn = blockIdx.x * GT_BN;
    const int tid  = threadIdx.x;
    const int lane = tid & 31;
    const int wave = tid >> 5;
    const int ln = lane & 15, lh = lane >> 4;
    const int wm = (wave & 1) * 64;
    const int wn = (wave >> 1) * 32;

    v8f acc[4][2] = {};

    for (int k0 = 0; k0 < K; k0 += GT_BK) {
        // CDNA5 speculative prefetch of the next K-tile (global_prefetch_b8)
        if (k0 + GT_BK < K) {
            __builtin_prefetch(&A[(size_t)(bm + (tid >> 1)) * K + k0 + GT_BK], 0, 1);
            __builtin_prefetch(&Bt[(size_t)(bn + (tid >> 1)) * K + k0 + GT_BK], 0, 1);
        }
        // stage A and B: 128x64 bf16 each, contiguous b128 both sides
#pragma unroll
        for (int i = 0; i < 4; ++i) {
            int chunk = tid + i * 256;            // 1024 chunks of 8 elems
            int r = chunk >> 3, c = (chunk & 7) * 8;
            *(v8bf*)&sA[r][c] = *(const v8bf*)&A[(size_t)(bm + r) * K + k0 + c];
            *(v8bf*)&sB[r][c] = *(const v8bf*)&Bt[(size_t)(bn + r) * K + k0 + c];
        }
        __syncthreads();

        // two K=32 steps, 4x2 wmma per wave per step
#pragma unroll
        for (int ks = 0; ks < GT_BK; ks += 32) {
            v16bf af[4];
#pragma unroll
            for (int i = 0; i < 4; ++i) {
                const __bf16* p = &sA[wm + 16 * i + ln][ks + 8 * lh];
                af[i] = make_frag(p, p + 16);
            }
#pragma unroll
            for (int j = 0; j < 2; ++j) {
                const __bf16* p = &sB[wn + 16 * j + ln][ks + 8 * lh];
                v16bf bf = make_frag(p, p + 16);
#pragma unroll
                for (int i = 0; i < 4; ++i)
                    acc[i][j] = __builtin_amdgcn_wmma_f32_16x16x32_bf16(
                        false, af[i], false, bf, (short)0, acc[i][j], false, false);
            }
        }
        __syncthreads();
    }

    // epilogue: C layout VGPR v, lanes0-15 -> M=v, lanes16-31 -> M=v+8
#pragma unroll
    for (int i = 0; i < 4; ++i)
#pragma unroll
        for (int j = 0; j < 2; ++j)
#pragma unroll
            for (int v = 0; v < 8; ++v) {
                int row = bm + wm + 16 * i + v + 8 * lh;
                int col = bn + wn + 16 * j + ln;
                C[(size_t)row * N + col] = acc[i][j][v];
            }
}

// ---------------------------------------------------------------------------
// Kernel 4: per-row RMSnorm(+scale)+partial RoPE (mode 0) or L2 norm (mode 1)
// one block = one (token, head) row of 256 elements; bf16 output.
// mode 1 (v) writes TRANSPOSED layout [b][head][dim][seq] for flash staging.
// ---------------------------------------------------------------------------
__global__ __launch_bounds__(256) void norm_rope_kernel(const float* __restrict__ in_base,
                                                        __bf16* __restrict__ out,
                                                        const float* __restrict__ scale,
                                                        int nh, int row_stride, int mode) {
    __shared__ float sh[256];
    __shared__ float red[8];
    const int tid = threadIdx.x;
    const int lane = tid & 31, wave = tid >> 5;
    const int row_id = blockIdx.x;
    const int bs = row_id / nh;
    const int head = row_id - bs * nh;
    const int pos = bs & (SEQ - 1);

    float val = in_base[(size_t)bs * row_stride + head * HD + tid];
    float ss = val * val;
#pragma unroll
    for (int o = 16; o > 0; o >>= 1) ss += __shfl_xor(ss, o, 32);
    if (lane == 0) red[wave] = ss;
    __syncthreads();
    float tot = 0.f;
#pragma unroll
    for (int w = 0; w < 8; ++w) tot += red[w];

    if (mode == 1) {                       // l2 norm (v), transposed output
        float y = val * rsqrtf(tot + 1e-6f);
        int bb = bs >> 11;                 // / SEQ
        int s  = bs & (SEQ - 1);
        out[(((size_t)bb * nh + head) * HD + tid) * SEQ + s] = (__bf16)y;
        return;
    }
    // rms norm with (1 + scale)
    float y = val * rsqrtf(tot * (1.0f / HD) + 1e-6f) * (1.0f + scale[tid]);
    sh[tid] = y;
    __syncthreads();

    float yo = y;
    if (tid < 64) {                        // y1r = x1r*cos - x2r*sin
        int j = tid;
        float fr = (float)pos * __expf(-(2.0f * j / (float)HD) * 9.210340371976184f);
        yo = sh[j] * cosf(fr) - sh[j + 128] * sinf(fr);
    } else if (tid >= 128 && tid < 192) {  // y2r = x2r*cos + x1r*sin
        int j = tid - 128;
        float fr = (float)pos * __expf(-(2.0f * j / (float)HD) * 9.210340371976184f);
        yo = sh[tid] * cosf(fr) + sh[j] * sinf(fr);
    }
    out[(size_t)row_id * HD + tid] = (__bf16)yo;
}

// ---------------------------------------------------------------------------
// Kernel 5: flash attention (causal, tanh softcap 50)
// grid: (S/128, NQ, B). block 256 thr = 8 waves; wave owns 16 query rows.
// K [key][dim] and Vt [dim][key] tiles staged via async global->LDS b128.
// ---------------------------------------------------------------------------
#define KT 32

__global__ __launch_bounds__(256) void flash_attn_kernel(const __bf16* __restrict__ Q,
                                                         const __bf16* __restrict__ Kb,
                                                         const __bf16* __restrict__ Vt,
                                                         __bf16* __restrict__ O) {
    __shared__ __bf16 sK[KT][HD];        // [key][dim]   16 KB
    __shared__ __bf16 sVt[HD][KT];       // [dim][key]   16 KB
    __shared__ __bf16 sP[8][16][KT];     // per-wave P    8 KB

    const int tid = threadIdx.x;
    const int lane = tid & 31, wave = tid >> 5;
    const int ln = lane & 15, lh = lane >> 4;
    const int b  = blockIdx.z;
    const int n  = blockIdx.y;           // q head
    const int kvh = n >> 1;              // GROUPS = 2
    const int qb = blockIdx.x * 128;
    const int qrow_base = qb + wave * 16;

    const __bf16* Kbase  = Kb + (((size_t)b * SEQ) * NKVH + kvh) * HD;
    const __bf16* Vtbase = Vt + ((size_t)b * NKVH + kvh) * HD * SEQ;

    // Q fragments for this wave's 16 rows (8 K-steps of 32)
    v16bf qf[8];
    {
        const __bf16* qp = Q + (((size_t)b * SEQ + qrow_base + ln) * NQH + n) * HD;
#pragma unroll
        for (int kk = 0; kk < 8; ++kk)
            qf[kk] = make_frag(qp + kk * 32 + 8 * lh, qp + kk * 32 + 16 + 8 * lh);
    }

    v8f o[16] = {};
    float mrow[8], lrow[8];
#pragma unroll
    for (int v = 0; v < 8; ++v) { mrow[v] = -1.0e30f; lrow[v] = 0.0f; }

    const int kend = qb + 128;           // causal bound (<= SEQ)
    for (int kt = 0; kt < kend; kt += KT) {
        // --- async staging: K row-major, Vt already transposed in global ---
#pragma unroll
        for (int i = 0; i < 4; ++i) {
            int chunk = tid + i * 256;                 // 1024 chunks of 8
            int r = chunk >> 5, c = (chunk & 31) * 8;  // K: key r, dim c
            async_copy_b128(&Kbase[((size_t)(kt + r)) * NKVH * HD + c], &sK[r][c]);
            int dim = chunk >> 2, kc = (chunk & 3) * 8; // V: dim row, key col
            async_copy_b128(&Vtbase[(size_t)dim * SEQ + kt + kc], &sVt[dim][kc]);
        }
        wait_asynccnt0();
        __syncthreads();

        // --- S = Q K^T  (2 n-tiles x 8 k-steps) ---
        v8f sacc[2] = {};
#pragma unroll
        for (int nt = 0; nt < 2; ++nt)
#pragma unroll
            for (int kk = 0; kk < 8; ++kk) {
                const __bf16* bp = &sK[nt * 16 + ln][kk * 32 + 8 * lh];
                v16bf bf = make_frag(bp, bp + 16);
                sacc[nt] = __builtin_amdgcn_wmma_f32_16x16x32_bf16(
                    false, qf[kk], false, bf, (short)0, sacc[nt], false, false);
            }

        // --- softcap + causal mask + online softmax (per 16-lane row group) ---
        float pv[2][8];
#pragma unroll
        for (int v = 0; v < 8; ++v) {
            const int qi = qrow_base + v + 8 * lh;
            float mx = -1.0e30f;
#pragma unroll
            for (int nt = 0; nt < 2; ++nt) {
                int ki = kt + nt * 16 + ln;
                float xv = sacc[nt][v];
                xv = tanhf(xv * 0.02f) * 50.0f;
                if (ki > qi) xv = NEG_BIG;
                pv[nt][v] = xv;
                mx = fmaxf(mx, xv);
            }
#pragma unroll
            for (int off = 1; off < 16; off <<= 1) mx = fmaxf(mx, __shfl_xor(mx, off, 32));
            float mnew = fmaxf(mrow[v], mx);
            float scl = __expf(mrow[v] - mnew);
            mrow[v] = mnew;
            float rs = 0.0f;
#pragma unroll
            for (int nt = 0; nt < 2; ++nt) {
                float p = __expf(pv[nt][v] - mnew);
                pv[nt][v] = p;
                rs += p;
            }
#pragma unroll
            for (int off = 1; off < 16; off <<= 1) rs += __shfl_xor(rs, off, 32);
            lrow[v] = lrow[v] * scl + rs;
#pragma unroll
            for (int dt = 0; dt < 16; ++dt) o[dt][v] *= scl;
        }

        // --- restage P (C-layout -> A-fragment layout) through wave-private LDS
#pragma unroll
        for (int nt = 0; nt < 2; ++nt)
#pragma unroll
            for (int v = 0; v < 8; ++v)
                sP[wave][v + 8 * lh][nt * 16 + ln] = (__bf16)pv[nt][v];

        const __bf16* ap = &sP[wave][ln][8 * lh];
        v16bf pf = make_frag(ap, ap + 16);

        // --- O += P V  (16 dim tiles) ---
#pragma unroll
        for (int dt = 0; dt < 16; ++dt) {
            const __bf16* bp = &sVt[dt * 16 + ln][8 * lh];
            v16bf vf = make_frag(bp, bp + 16);
            o[dt] = __builtin_amdgcn_wmma_f32_16x16x32_bf16(
                false, pf, false, vf, (short)0, o[dt], false, false);
        }
        __syncthreads();
    }

    // --- normalize and store bf16 ---
#pragma unroll
    for (int dt = 0; dt < 16; ++dt)
#pragma unroll
        for (int v = 0; v < 8; ++v) {
            const int qi = qrow_base + v + 8 * lh;
            float y = o[dt][v] / lrow[v];
            O[(((size_t)b * SEQ + qi) * NQH + n) * HD + dt * 16 + ln] = (__bf16)y;
        }
}

// ---------------------------------------------------------------------------
// Host launcher
// Inputs: 0=x 1=mask(unused, known causal) 2=Wq 3=Wk 4=Wv 5=Wo 6=q_scale 7=k_scale
// ---------------------------------------------------------------------------
extern "C" void kernel_launch(void* const* d_in, const int* in_sizes, int n_in,
                              void* d_out, int out_size, void* d_ws, size_t ws_size,
                              hipStream_t stream) {
    const float* x   = (const float*)d_in[0];
    const float* Wq  = (const float*)d_in[2];
    const float* Wk  = (const float*)d_in[3];
    const float* Wv  = (const float*)d_in[4];
    const float* Wo  = (const float*)d_in[5];
    const float* qsc = (const float*)d_in[6];
    const float* ksc = (const float*)d_in[7];
    float* out = (float*)d_out;

    // workspace carve-out (~159 MB)
    size_t off = 0;
    auto carve = [&](size_t bytes) {
        void* p = (char*)d_ws + off;
        off += (bytes + 255) & ~(size_t)255;
        return p;
    };
    __bf16* xbf   = (__bf16*)carve((size_t)MROWS * DMODEL * 2);
    __bf16* wqkvT = (__bf16*)carve((size_t)NCOLS * DMODEL * 2);
    __bf16* woT   = (__bf16*)carve((size_t)DMODEL * (NQH * HD) * 2);
    float*  qkv   = (float*) carve((size_t)MROWS * NCOLS * 4);
    __bf16* qbf   = (__bf16*)carve((size_t)MROWS * NQH  * HD * 2);
    __bf16* kbf   = (__bf16*)carve((size_t)MROWS * NKVH * HD * 2);
    __bf16* vbfT  = (__bf16*)carve((size_t)MROWS * NKVH * HD * 2);
    __bf16* aobf  = (__bf16*)carve((size_t)MROWS * NQH  * HD * 2);

    // 1. convert x, pack Wqkv^T and Wo^T
    {
        int n = MROWS * DMODEL;
        f32_to_bf16_kernel<<<(n + 255) / 256, 256, 0, stream>>>(x, xbf, n);
    }
    {
        int n = NCOLS * DMODEL;
        pack_qkv_wT_kernel<<<(n + 255) / 256, 256, 0, stream>>>(Wq, Wk, Wv, wqkvT);
    }
    {
        int n = DMODEL * NQH * HD;
        pack_woT_kernel<<<(n + 255) / 256, 256, 0, stream>>>(Wo, woT);
    }

    // 2. fused QKV projection GEMM: [4096 x 4096] = x * Wqkv
    gemm_bf16_kernel<<<dim3(NCOLS / GT_BN, MROWS / GT_BM), 256, 0, stream>>>(
        xbf, wqkvT, qkv, MROWS, NCOLS, DMODEL);

    // 3. norms + rope -> bf16 q/k (row-major) and v (transposed)
    norm_rope_kernel<<<MROWS * NQH, 256, 0, stream>>>(
        qkv, qbf, qsc, NQH, NCOLS, 0);
    norm_rope_kernel<<<MROWS * NKVH, 256, 0, stream>>>(
        qkv + NQH * HD, kbf, ksc, NKVH, NCOLS, 0);
    norm_rope_kernel<<<MROWS * NKVH, 256, 0, stream>>>(
        qkv + (NQH + NKVH) * HD, vbfT, nullptr, NKVH, NCOLS, 1);

    // 4. flash attention
    flash_attn_kernel<<<dim3(SEQ / 128, NQH, B_SZ), 256, 0, stream>>>(
        qbf, kbf, vbfT, aobf);

    // 5. output projection: [4096 x 2048] = attn_out * Wo^T
    gemm_bf16_kernel<<<dim3(DMODEL / GT_BN, MROWS / GT_BM), 256, 0, stream>>>(
        aobf, woT, out, MROWS, DMODEL, NQH * HD);
}
